// TriangleMultiplication_64321430225396
// MI455X (gfx1250) — matrix operand order repaired and checked by reading, not compiled, not run on hardware
//
#include <hip/hip_runtime.h>
#include <hip/hip_bf16.h>

typedef __bf16 bf16;
typedef __bf16 v16bf __attribute__((ext_vector_type(16)));
typedef float  v8f   __attribute__((ext_vector_type(8)));
typedef unsigned int u32x4 __attribute__((ext_vector_type(4)));
typedef int          i32x4 __attribute__((ext_vector_type(4)));
typedef int          i32x8 __attribute__((ext_vector_type(8)));

#define DIM 128
#define LL  512
#define NTOT (LL * LL)   // 262144 rows

struct U8x32 { uint4 a, b; };   // 32 bytes -> v16bf via bit_cast

static __device__ __forceinline__ bf16 f2bf(float f) { return (bf16)f; }

static __device__ __forceinline__ v16bf load_frag_g(const bf16* p0, const bf16* p1) {
  // two 16B chunks (8 halfs each) -> one v16bf operand
  U8x32 u;
  u.a = *reinterpret_cast<const uint4*>(p0);
  u.b = *reinterpret_cast<const uint4*>(p1);
  return __builtin_bit_cast(v16bf, u);
}

// ---------------------------------------------------------------------------
// Kernel 1: fused projections  left/right/gate = act(pair @ W + b), bf16 out.
// One wave per 16-row n-tile; weights staged transposed in LDS ([h][k]) so
// B fragments are contiguous ds_load_b128s.
// ---------------------------------------------------------------------------
__global__ __launch_bounds__(256) void k_proj(
    const float* __restrict__ pair,
    const float* __restrict__ Wl, const float* __restrict__ bl,
    const float* __restrict__ Wr, const float* __restrict__ br,
    const float* __restrict__ Wg, const float* __restrict__ bg,
    bf16* __restrict__ left, bf16* __restrict__ right, bf16* __restrict__ gate)
{
  __shared__ bf16 wT[DIM * DIM];   // transposed weights: wT[h*128 + k]

  const int wave = threadIdx.x >> 5;
  const int lane = threadIdx.x & 31;
  const int cn = lane & 15;        // column-in-tile (N) / row-in-tile for A
  const int hf = lane >> 4;        // half-wave selector
  const long n0 = ((long)blockIdx.x * 8 + wave) * 16;

  // ---- A fragments: pair rows n0..n0+15, K = 0..127, converted to bf16 ----
  v16bf A[4];
  const float* prow = pair + (n0 + cn) * DIM;
#pragma unroll
  for (int kk = 0; kk < 4; ++kk) {
    const int kb = kk * 32 + hf * 8;   // halfs 0..7 -> K kb..kb+7 ; 8..15 -> +16
    v16bf a;
#pragma unroll
    for (int t = 0; t < 8; ++t) a[t]     = f2bf(prow[kb + t]);
#pragma unroll
    for (int t = 0; t < 8; ++t) a[8 + t] = f2bf(prow[kb + 16 + t]);
    A[kk] = a;
  }

#pragma unroll
  for (int m = 0; m < 3; ++m) {
    const float* W    = (m == 0) ? Wl : (m == 1) ? Wr : Wg;
    const float* bias = (m == 0) ? bl : (m == 1) ? br : bg;
    bf16*        dst  = (m == 0) ? left : (m == 1) ? right : gate;

    __syncthreads();
    for (int idx = threadIdx.x; idx < DIM * DIM; idx += 256) {
      const int k = idx / DIM, h = idx % DIM;     // coalesced read over h
      wT[h * DIM + k] = f2bf(W[idx]);
    }
    __syncthreads();

#pragma unroll
    for (int ht = 0; ht < 8; ++ht) {
      const int h = ht * 16 + cn;
      v8f acc = {};
#pragma unroll
      for (int kk = 0; kk < 4; ++kk) {
        const bf16* bp = &wT[h * DIM + kk * 32 + hf * 8];   // halfs 0..7 -> K kb..kb+7 ; 8..15 -> +16
        v16bf B = load_frag_g(bp, bp + 16);
        acc = __builtin_amdgcn_wmma_f32_16x16x32_bf16(
            false, A[kk], false, B, (short)0, acc, false, false);
      }
      const float bv = bias[h];
#pragma unroll
      for (int r = 0; r < 8; ++r) {
        float v = acc[r] + bv;
        if (m == 2) v = 1.0f / (1.0f + __expf(-v));   // sigmoid gate
        dst[(n0 + r + hf * 8) * DIM + h] = f2bf(v);
      }
    }
  }
}

// ---------------------------------------------------------------------------
// Generic strided bf16 transpose (32x32 LDS tiles), used to build K-contiguous
// layouts for the WMMA GEMMs.  src = b*srcBS + r*srcRS + c ; dst = b*dstBS +
// c*dstRS + r.  All dims multiples of 32.
// ---------------------------------------------------------------------------
__global__ __launch_bounds__(256) void k_transpose(
    const bf16* __restrict__ src, bf16* __restrict__ dst,
    long srcRS, long srcBS, long dstRS, long dstBS)
{
  __shared__ bf16 tile[32][33];
  const long b  = blockIdx.z;
  const long r0 = (long)blockIdx.y * 32;
  const long c0 = (long)blockIdx.x * 32;
  const int tx = threadIdx.x & 31;
  const int ty = threadIdx.x >> 5;
#pragma unroll
  for (int p = 0; p < 4; ++p) {
    const int r = ty + p * 8;
    tile[r][tx] = src[b * srcBS + (r0 + r) * srcRS + (c0 + tx)];
  }
  __syncthreads();
#pragma unroll
  for (int p = 0; p < 4; ++p) {
    const int c = ty + p * 8;
    dst[b * dstBS + (c0 + c) * dstRS + (r0 + tx)] = tile[tx][c];
  }
}

// ---------------------------------------------------------------------------
// Kernel 3: triangle einsum as 128 batched 512x512x512 bf16 GEMMs.
//   C_h[i][j] = sum_k Lt[h][i][k] * Rt[h][j][k]
// TDM double-buffered LDS staging (tensor_load_to_lds + s_wait_tensorcnt),
// fragments via contiguous LDS b128 reads, WMMA bf16 accumulation.
// Block: 256 thr = 8 waves, tile 64(i) x 64(j) per h.
// ---------------------------------------------------------------------------
static __device__ __forceinline__ void tdm_issue_2d(
    unsigned long gaddr, unsigned lds_addr)
{
  // D# group 0: count=1, lds_addr, global_addr[56:0], type=2
  u32x4 g0;
  g0[0] = 1u;
  g0[1] = lds_addr;
  g0[2] = (unsigned)(gaddr & 0xffffffffu);
  g0[3] = (unsigned)((gaddr >> 32) & 0x01ffffffu) | (2u << 30);
  // D# group 1: data_size=2B, tensor 512x512, tile 32(k) x 64(rows),
  // dim0_stride = 512 elements.
  i32x8 g1;
  g1[0] = (int)(1u << 16);        // workgroup_mask=0 | data_size=1 (2 bytes)
  g1[1] = (int)(512u << 16);      // tensor_dim0[15:0] = 512
  g1[2] = (int)(512u << 16);      // tensor_dim0[31:16]=0 | tensor_dim1[15:0]=512
  g1[3] = (int)(32u << 16);       // tensor_dim1[31:16]=0 | tile_dim0 = 32
  g1[4] = 64;                     // tile_dim1 = 64 | tile_dim2 = 0
  g1[5] = 512;                    // tensor_dim0_stride[31:0] = 512
  g1[6] = 0;                      // stride hi | dim1_stride lo
  g1[7] = 0;
  i32x4 g2 = {0, 0, 0, 0};
  i32x4 g3 = {0, 0, 0, 0};
#if __clang_major__ >= 23
  i32x8 gz = {0, 0, 0, 0, 0, 0, 0, 0};
  __builtin_amdgcn_tensor_load_to_lds(g0, g1, g2, g3, gz, 0);
#else
  __builtin_amdgcn_tensor_load_to_lds(g0, g1, g2, g3, 0);
#endif
}

__global__ __launch_bounds__(256) void k_tri(
    const bf16* __restrict__ Lt,   // [h][i][k]
    const bf16* __restrict__ Rt,   // [h][j][k]
    bf16* __restrict__ Ct)         // [h][i][j]
{
  __shared__ bf16 bufA[2][64 * 32];
  __shared__ bf16 bufB[2][64 * 32];

  const int h  = blockIdx.z;
  const int i0 = blockIdx.y * 64;
  const int j0 = blockIdx.x * 64;
  const int wave = threadIdx.x >> 5;
  const int lane = threadIdx.x & 31;
  const int cn = lane & 15, hf = lane >> 4;
  const int isub = wave & 3;        // 4 i-subtiles of 16
  const int jh   = wave >> 1 & 0;   // (unused placeholder removed below)
  const int jhalf = wave >> 2;      // waves 0-3: j 0..31, waves 4-7: j 32..63

  const unsigned long baseL = ((unsigned long)h * NTOT + (unsigned long)i0 * 512);
  const unsigned long baseR = ((unsigned long)h * NTOT + (unsigned long)j0 * 512);
  const unsigned ldsA[2] = {(unsigned)(size_t)&bufA[0][0], (unsigned)(size_t)&bufA[1][0]};
  const unsigned ldsB[2] = {(unsigned)(size_t)&bufB[0][0], (unsigned)(size_t)&bufB[1][0]};
  (void)jh;

  v8f acc[2] = {v8f{}, v8f{}};

  if (wave == 0) {   // prologue: fill buffer 0
    tdm_issue_2d((unsigned long)(size_t)(Lt + baseL), ldsA[0]);
    tdm_issue_2d((unsigned long)(size_t)(Rt + baseR), ldsB[0]);
    __builtin_amdgcn_s_wait_tensorcnt(0);
  }
  __syncthreads();

  for (int it = 0; it < 16; ++it) {           // K loop, 32 per step
    const int buf = it & 1;
    if (wave == 0 && it + 1 < 16) {           // async prefetch next chunk
      const unsigned long kc = (unsigned long)(it + 1) * 32;
      tdm_issue_2d((unsigned long)(size_t)(Lt + baseL + kc), ldsA[buf ^ 1]);
      tdm_issue_2d((unsigned long)(size_t)(Rt + baseR + kc), ldsB[buf ^ 1]);
    }

    // A fragment: rows i = isub*16 + cn, K split across half-waves
    const bf16* ap = &bufA[buf][(isub * 16 + cn) * 32 + hf * 8];
    v16bf Af = load_frag_g(ap, ap + 16);
#pragma unroll
    for (int t = 0; t < 2; ++t) {
      const int jsub = jhalf * 2 + t;
      const bf16* bp = &bufB[buf][(jsub * 16 + cn) * 32 + hf * 8];
      v16bf Bf = load_frag_g(bp, bp + 16);
      acc[t] = __builtin_amdgcn_wmma_f32_16x16x32_bf16(
          false, Af, false, Bf, (short)0, acc[t], false, false);
    }

    if (wave == 0 && it + 1 < 16) __builtin_amdgcn_s_wait_tensorcnt(0);
    __syncthreads();
  }

#pragma unroll
  for (int t = 0; t < 2; ++t) {
    const int jsub = jhalf * 2 + t;
#pragma unroll
    for (int r = 0; r < 8; ++r) {
      const long addr = (long)h * NTOT +
                        (long)(i0 + isub * 16 + r + hf * 8) * 512 +
                        (j0 + jsub * 16 + cn);
      Ct[addr] = f2bf(acc[t][r]);
    }
  }
}

// ---------------------------------------------------------------------------
// Kernel 4: out = LayerNorm(pair + gate * (tri @ Wo + bo)) * gamma + beta
// One wave per 16-row tile; Wo staged transposed in LDS; LayerNorm via
// half-wave (16-lane) shfl_xor reductions per the WMMA C layout.
// ---------------------------------------------------------------------------
__global__ __launch_bounds__(256) void k_final(
    const bf16* __restrict__ tri2,   // [n][h] bf16
    const float* __restrict__ pair,
    const bf16* __restrict__ gate,
    const float* __restrict__ Wo, const float* __restrict__ bo,
    const float* __restrict__ gamma, const float* __restrict__ beta,
    float* __restrict__ out)
{
  __shared__ bf16 wT[DIM * DIM];   // WoT[d][h]
  for (int idx = threadIdx.x; idx < DIM * DIM; idx += 256) {
    const int hk = idx / DIM, d = idx % DIM;
    wT[d * DIM + hk] = f2bf(Wo[idx]);
  }
  __syncthreads();

  const int wave = threadIdx.x >> 5;
  const int lane = threadIdx.x & 31;
  const int cn = lane & 15, hf = lane >> 4;
  const long n0 = ((long)blockIdx.x * 8 + wave) * 16;

  // A fragments from tri2 (bf16, K = h contiguous)
  v16bf A[4];
  const bf16* arow = tri2 + (n0 + cn) * DIM;
#pragma unroll
  for (int kk = 0; kk < 4; ++kk) {
    const bf16* p = arow + kk * 32 + hf * 8;
    A[kk] = load_frag_g(p, p + 16);
  }

  float x[8][8];   // [d-tile][row-vgpr]
#pragma unroll
  for (int ht = 0; ht < 8; ++ht) {
    v8f acc = {};
#pragma unroll
    for (int kk = 0; kk < 4; ++kk) {
      const bf16* bp = &wT[(ht * 16 + cn) * DIM + kk * 32 + hf * 8];
      v16bf B = load_frag_g(bp, bp + 16);
      acc = __builtin_amdgcn_wmma_f32_16x16x32_bf16(
          false, A[kk], false, B, (short)0, acc, false, false);
    }
    const float bv = bo[ht * 16 + cn];
#pragma unroll
    for (int r = 0; r < 8; ++r) {
      const long off = (n0 + r + hf * 8) * DIM + ht * 16 + cn;
      const float g = (float)gate[off];
      x[ht][r] = pair[off] + g * (acc[r] + bv);
    }
  }

  // LayerNorm: each row's 128 values live on 16 lanes x 8 d-tiles
#pragma unroll
  for (int r = 0; r < 8; ++r) {
    float s = 0.f, s2 = 0.f;
#pragma unroll
    for (int ht = 0; ht < 8; ++ht) { s += x[ht][r]; s2 += x[ht][r] * x[ht][r]; }
#pragma unroll
    for (int m = 1; m < 16; m <<= 1) {   // stays within the 16-lane half-wave
      s  += __shfl_xor(s, m, 32);
      s2 += __shfl_xor(s2, m, 32);
    }
    const float mu  = s * (1.0f / DIM);
    const float var = s2 * (1.0f / DIM) - mu * mu;
    const float rs  = __frsqrt_rn(var + 1e-5f);
    const long n = n0 + r + hf * 8;
#pragma unroll
    for (int ht = 0; ht < 8; ++ht) {
      const int hc = ht * 16 + cn;
      out[n * DIM + hc] = (x[ht][r] - mu) * rs * gamma[hc] + beta[hc];
    }
  }
}

// ---------------------------------------------------------------------------
extern "C" void kernel_launch(void* const* d_in, const int* in_sizes, int n_in,
                              void* d_out, int out_size, void* d_ws, size_t ws_size,
                              hipStream_t stream) {
  (void)in_sizes; (void)n_in; (void)out_size; (void)ws_size;
  const float* pair  = (const float*)d_in[0];
  const float* Wl    = (const float*)d_in[1];
  const float* bl    = (const float*)d_in[2];
  const float* Wr    = (const float*)d_in[3];
  const float* br    = (const float*)d_in[4];
  const float* Wo    = (const float*)d_in[5];
  const float* bo    = (const float*)d_in[6];
  const float* Wg    = (const float*)d_in[7];
  const float* bg    = (const float*)d_in[8];
  const float* gamma = (const float*)d_in[9];
  const float* beta  = (const float*)d_in[10];
  float* out = (float*)d_out;

  const size_t SZ = (size_t)NTOT * DIM * sizeof(bf16);  // 64 MB per buffer
  char* w = (char*)d_ws;
  bf16* left   = (bf16*)(w + 0 * SZ);   // [n][h]
  bf16* right  = (bf16*)(w + 1 * SZ);   // [n][h]
  bf16* gate   = (bf16*)(w + 2 * SZ);   // [n][h]
  bf16* leftT  = (bf16*)(w + 3 * SZ);   // [h][i][k]
  bf16* rightT = (bf16*)(w + 4 * SZ);   // [h][j][k]
  bf16* triT   = left;                  // reuse: [h][i][j] (left dead after transpose)
  bf16* tri2   = right;                 // reuse: [n][h]    (right dead after transpose)

  // 1) projections (WMMA bf16)
  k_proj<<<2048, 256, 0, stream>>>(pair, Wl, bl, Wr, br, Wg, bg, left, right, gate);

  // 2) build K-contiguous layouts
  // left [n][h] -> leftT [h][n]           (n = i*512 + k)
  k_transpose<<<dim3(DIM / 32, NTOT / 32, 1), 256, 0, stream>>>(
      left, leftT, /*srcRS*/ DIM, 0, /*dstRS*/ (long)NTOT, 0);
  // right [k][j][h] -> rightT [h][j][k]   (batched over j)
  k_transpose<<<dim3(DIM / 32, LL / 32, LL), 256, 0, stream>>>(
      right, rightT, /*srcRS*/ (long)LL * DIM, /*srcBS*/ DIM,
      /*dstRS*/ (long)NTOT, /*dstBS*/ LL);

  // 3) triangle einsum: 128 batched bf16 GEMMs (TDM staging + WMMA)
  k_tri<<<dim3(LL / 64, LL / 64, DIM), 256, 0, stream>>>(leftT, rightT, triT);

  // 3b) triT [h][n] -> tri2 [n][h]
  k_transpose<<<dim3(NTOT / 32, DIM / 32, 1), 256, 0, stream>>>(
      triT, tri2, /*srcRS*/ (long)NTOT, 0, /*dstRS*/ DIM, 0);

  // 4) output GEMM + gated residual + LayerNorm (WMMA bf16 + wave32 shfl)
  k_final<<<2048, 256, 0, stream>>>(tri2, pair, gate, Wo, bo, gamma, beta, out);
}
